// Context_Encoder_47201690583383
// MI455X (gfx1250) — compile-verified, matrix operand order checked by reading, not run on hardware
//
#include <hip/hip_runtime.h>
#include <hip/hip_bf16.h>
#include <stdint.h>

// ---------------- problem constants ----------------
#define B_    32
#define S_    512
#define D_    512
#define H2_   256
#define G3_   768          // 3*H2
#define HEADS_ 8
#define DK_   64
#define BS_   (B_*S_)      // 16384
#define LN_EPS_ 1e-5f

typedef __attribute__((ext_vector_type(16))) _Float16 v16h;
typedef __attribute__((ext_vector_type(8)))  float    v8f;
typedef __attribute__((ext_vector_type(4)))  unsigned int v4u;
typedef __attribute__((ext_vector_type(8)))  int      v8i;
typedef __attribute__((ext_vector_type(4)))  int      v4i;

#define DEV static __device__ __forceinline__

#if defined(__gfx1250__) && __has_builtin(__builtin_amdgcn_tensor_load_to_lds)
#define HAVE_TDM 1
#else
#define HAVE_TDM 0
#endif

DEV void wait_tensorcnt0() {
#if __has_builtin(__builtin_amdgcn_s_wait_tensorcnt)
    __builtin_amdgcn_s_wait_tensorcnt(0);
#else
    asm volatile("s_wait_tensorcnt 0x0" ::: "memory");
#endif
}

#if HAVE_TDM
// One-shot 2D tile DMA: global (row-major, stride_elems f16/row) -> LDS at
// byte offset lds_off, with LDS padding of 4 DWORDs after every
// (2<<pad_interval) DWORDs stored (ISA 8.4: pad threshold = 2<<pad_interval DW,
// pad_amount field = N-1 DW). data_size=2 bytes. 2-D tile: groups 2/3 zero.
DEV void tdm_load_2d(const void* gaddr, unsigned lds_off,
                     unsigned tile0_elems, unsigned tile1_rows,
                     unsigned stride_elems, unsigned pad_interval) {
    unsigned long long ga = (unsigned long long)(uintptr_t)gaddr;
    const unsigned dim1 = 0x40000000u;                     // large tensor_dim1 (OOB bound)
    v4u g0;
    g0[0] = 1u;                                            // count=1 (valid), user mode
    g0[1] = lds_off;                                       // lds_addr (bytes)
    g0[2] = (unsigned)(ga & 0xFFFFFFFFu);                  // global_addr[31:0]
    g0[3] = (unsigned)((ga >> 32) & 0x01FFFFFFu) | (2u << 30); // addr[56:32] | type=2
    v8i g1;
    g1[0] = (int)((1u << 16)                               // data_size = 2 bytes
                | (1u << 20)                               // pad_enable
                | (pad_interval << 22)                     // pad every 2<<pi DWORDs
                | (3u << 25));                             // pad_amount = 4 DWORDs
    g1[1] = (int)((stride_elems & 0xFFFFu) << 16);         // tensor_dim0[15:0] @ bit48
    g1[2] = (int)(((stride_elems >> 16) & 0xFFFFu)         // tensor_dim0[31:16]
                | ((dim1 & 0xFFFFu) << 16));               // tensor_dim1[15:0]
    g1[3] = (int)(((dim1 >> 16) & 0xFFFFu)                 // tensor_dim1[31:16]
                | (tile0_elems << 16));                    // tile_dim0
    g1[4] = (int)(tile1_rows & 0xFFFFu);                   // tile_dim1 (tile_dim2=0)
    g1[5] = (int)(stride_elems);                           // tensor_dim0_stride[31:0]
    g1[6] = 0;                                             // stride[47:32] | dim1_stride lo
    g1[7] = 0;
    v4i z4 = {0, 0, 0, 0};
#if __clang_major__ >= 23
    v8i z8 = {0, 0, 0, 0, 0, 0, 0, 0};
    __builtin_amdgcn_tensor_load_to_lds(g0, g1, z4, z4, z8, 0);
#else
    __builtin_amdgcn_tensor_load_to_lds(g0, g1, z4, z4, 0);
#endif
}
#endif // HAVE_TDM

// Load a 16-half WMMA fragment slice (two b128 reads), 16B-aligned.
DEV v16h ldfrag(const _Float16* p) {
    v16h r;
    ((uint4*)&r)[0] = *(const uint4*)(p);
    ((uint4*)&r)[1] = *(const uint4*)(p + 8);
    return r;
}

DEV v8f wmma_f16(v16h a, v16h b, v8f c) {
    // D = A(16x32,f16) * B(32x16,f16) + C(16x16,f32)
    return __builtin_amdgcn_wmma_f32_16x16x32_f16(
        false, a, false, b, (short)0, c, false, false);
}

// block=128 sum reduction (4 waves)
DEV float blk_sum128(float v, float* red) {
    #pragma unroll
    for (int o = 16; o; o >>= 1) v += __shfl_xor(v, o, 32);
    __syncthreads();
    if ((threadIdx.x & 31) == 0) red[threadIdx.x >> 5] = v;
    __syncthreads();
    return red[0] + red[1] + red[2] + red[3];
}

// ---------------- f32 -> f16 cast ----------------
__global__ void cast16(const float* __restrict__ s, _Float16* __restrict__ d, int n) {
    int i = blockIdx.x * blockDim.x + threadIdx.x;
    int stride = gridDim.x * blockDim.x;
    for (; i < n; i += stride) d[i] = (_Float16)s[i];
}

// ---------------- generic WMMA GEMM: C[M,N] = A[M,K] * W[N,K]^T + bias ----------------
// block 256 = 8 waves, tile 128x64, each wave does a 32x32 patch (2x2 WMMA tiles).
// Tiles staged into LDS by the Tensor Data Mover (one descriptor per tile).
template<bool OUTF16>
__global__ __launch_bounds__(256) void gemm128x64(const _Float16* __restrict__ A,
                                                  const _Float16* __restrict__ W,
                                                  const float* __restrict__ bias,
                                                  void* __restrict__ out,
                                                  int K, int N) {
    __shared__ _Float16 smem[128 * 40 + 64 * 40];
    _Float16 (*As)[40] = (_Float16(*)[40])smem;            // LDS byte offset 0
    _Float16 (*Ws)[40] = (_Float16(*)[40])(smem + 128 * 40); // LDS byte offset 10240
    const int tid  = threadIdx.x;
    const int lane = tid & 31, wave = tid >> 5;
    const int wm = wave >> 1, wn = wave & 1;
    const int ll = lane & 15, lh = lane >> 4;
    const size_t rowA0 = (size_t)blockIdx.x * 128;
    const int col0 = blockIdx.y * 64;

    v8f acc[2][2] = {};
    const int kblocks = K >> 5;
    for (int kb = 0; kb < kblocks; ++kb) {
#if HAVE_TDM
        if (wave == 0) {
            // A tile: 128 rows x 32 halves (64B row = 16 DW -> pad_interval=3)
            tdm_load_2d(&A[rowA0 * K + kb * 32], 0u, 32u, 128u, (unsigned)K, 3u);
            // W tile: 64 rows x 32 halves
            tdm_load_2d(&W[(size_t)col0 * K + kb * 32], 128u * 40u * 2u, 32u, 64u,
                        (unsigned)K, 3u);
            wait_tensorcnt0();
        }
#else
        #pragma unroll
        for (int i = 0; i < 2; ++i) {                 // A tile: 128x32 halves
            int t = tid + i * 256;
            int r = t >> 2, c = (t & 3) * 8;
            *(uint4*)&As[r][c] = *(const uint4*)&A[(rowA0 + r) * K + kb * 32 + c];
        }
        {                                             // W tile: 64x32 halves
            int r = tid >> 2, c = (tid & 3) * 8;
            *(uint4*)&Ws[r][c] = *(const uint4*)&W[(size_t)(col0 + r) * K + kb * 32 + c];
        }
        if (kb + 1 < kblocks)
            __builtin_prefetch(&A[(rowA0 + (tid >> 2)) * K + (kb + 1) * 32], 0, 1);
#endif
        __syncthreads();

        v16h af0 = ldfrag(&As[wm * 32 +      ll][lh * 16]);
        v16h af1 = ldfrag(&As[wm * 32 + 16 + ll][lh * 16]);
        v16h bf0 = ldfrag(&Ws[wn * 32 +      ll][lh * 16]);
        v16h bf1 = ldfrag(&Ws[wn * 32 + 16 + ll][lh * 16]);
        acc[0][0] = wmma_f16(af0, bf0, acc[0][0]);
        acc[0][1] = wmma_f16(af0, bf1, acc[0][1]);
        acc[1][0] = wmma_f16(af1, bf0, acc[1][0]);
        acc[1][1] = wmma_f16(af1, bf1, acc[1][1]);
        __syncthreads();
    }
    // epilogue: C layout  row = r + 8*(lane>=16), col = lane&15
    #pragma unroll
    for (int mi = 0; mi < 2; ++mi)
        #pragma unroll
        for (int ni = 0; ni < 2; ++ni)
            #pragma unroll
            for (int r = 0; r < 8; ++r) {
                size_t row = rowA0 + wm * 32 + mi * 16 + r + 8 * lh;
                int    col = col0 + wn * 32 + ni * 16 + ll;
                float v = acc[mi][ni][r] + bias[col];
                if (OUTF16) ((_Float16*)out)[row * (size_t)N + col] = (_Float16)v;
                else        ((float*)out)[row * (size_t)N + col] = v;
            }
}

// ---------------- persistent masked GRU scan ----------------
// grid = 4 workgroups: (dir in {fwd,bwd}) x (batch half of 16 rows).
// Per step: h16(16x256) @ w_hh^T(768x256) via WMMA; w_hh fragments are hoisted
// by the compiler into VGPRs across all 512 steps. Fused gate update in LDS.
__global__ __launch_bounds__(512) void gru_scan(const _Float16* __restrict__ xg_f,
                                                const _Float16* __restrict__ xg_b,
                                                const _Float16* __restrict__ whh_f,
                                                const _Float16* __restrict__ whh_b,
                                                const float* __restrict__ bhh_f,
                                                const float* __restrict__ bhh_b,
                                                const int* __restrict__ masks,
                                                const int* __restrict__ lengths,
                                                float* __restrict__ gru_out,
                                                float* __restrict__ out) {
    __shared__ float    h32[16][256];
    __shared__ _Float16 h16[16][264];
    __shared__ _Float16 hg16[16][776];

    const int tid = threadIdx.x;
    const int lane = tid & 31, wave = tid >> 5;
    const int ll = lane & 15, lh = lane >> 4;
    const int dir = blockIdx.x >> 1, bh = blockIdx.x & 1;
    const int bbase = bh * 16;
    const _Float16* xg  = dir ? xg_b  : xg_f;
    const _Float16* whh = dir ? whh_b : whh_f;
    const float*    bhh = dir ? bhh_b : bhh_f;

    #pragma unroll
    for (int i = 0; i < 8; ++i) { int e = i * 512 + tid; h32[e >> 8][e & 255] = 0.f; }
    __syncthreads();

    for (int t = 0; t < S_; ++t) {
        const int pos = dir ? (S_ - 1 - t) : t;
        #pragma unroll
        for (int i = 0; i < 8; ++i) { int e = i * 512 + tid; h16[e >> 8][e & 255] = (_Float16)h32[e >> 8][e & 255]; }
        __syncthreads();

        {   // hg = h @ whh^T ; wave owns N-tiles {3w,3w+1,3w+2} of 48
            v8f acc[3] = {};
            #pragma unroll
            for (int kb = 0; kb < 8; ++kb) {
                v16h a = ldfrag(&h16[ll][kb * 32 + lh * 16]);
                #pragma unroll
                for (int j = 0; j < 3; ++j) {
                    int nt = wave * 3 + j;
                    v16h b = ldfrag(&whh[(size_t)(nt * 16 + ll) * H2_ + kb * 32 + lh * 16]);
                    acc[j] = wmma_f16(a, b, acc[j]);
                }
            }
            #pragma unroll
            for (int j = 0; j < 3; ++j) {
                int nt = wave * 3 + j;
                #pragma unroll
                for (int r = 0; r < 8; ++r)
                    hg16[r + 8 * lh][nt * 16 + ll] = (_Float16)acc[j][r];
            }
        }
        __syncthreads();

        #pragma unroll
        for (int i = 0; i < 8; ++i) {
            int e = i * 512 + tid;
            int bl = e >> 8, j = e & 255;
            int gb = bbase + bl;
            float m = (float)masks[gb * S_ + pos];
            size_t xoff = ((size_t)gb * S_ + pos) * G3_;
            float xr = (float)xg[xoff + j];
            float xz = (float)xg[xoff + 256 + j];
            float xn = (float)xg[xoff + 512 + j];
            float hr = (float)hg16[bl][j]         + bhh[j];
            float hz = (float)hg16[bl][256 + j]   + bhh[256 + j];
            float hn = (float)hg16[bl][512 + j]   + bhh[512 + j];
            float rg = 1.f / (1.f + __expf(-(xr + hr)));
            float zg = 1.f / (1.f + __expf(-(xz + hz)));
            float ng = tanhf(xn + rg * hn);
            float h  = h32[bl][j];
            float hnew = (1.f - zg) * ng + zg * h;
            h = m * hnew + (1.f - m) * h;
            h32[bl][j] = h;
            gru_out[((size_t)gb * S_ + pos) * D_ + dir * H2_ + j] = h * m;
        }
        __syncthreads();
    }
    #pragma unroll
    for (int i = 0; i < 8; ++i) {
        int e = i * 512 + tid; int bl = e >> 8, j = e & 255; int gb = bbase + bl;
        float zm = lengths[gb] != 0 ? 1.f : 0.f;
        out[(size_t)BS_ * D_ + (size_t)gb * D_ + dir * H2_ + j] = h32[bl][j] * zm;
    }
}

// ---------------- LayerNorm over D=512, writes f32 + f16 ----------------
__global__ __launch_bounds__(128) void ln_write(const float* __restrict__ X,
                                                const float* __restrict__ g,
                                                const float* __restrict__ bb,
                                                float* __restrict__ Y32,
                                                _Float16* __restrict__ Y16) {
    __shared__ float red[4];
    size_t row = blockIdx.x;
    const float* x = X + row * D_;
    float v[4], s1 = 0.f, s2 = 0.f;
    #pragma unroll
    for (int i = 0; i < 4; ++i) { v[i] = x[threadIdx.x + i * 128]; s1 += v[i]; s2 += v[i] * v[i]; }
    float t1 = blk_sum128(s1, red);
    float t2 = blk_sum128(s2, red);
    float mu  = t1 * (1.f / D_);
    float inv = rsqrtf(fmaxf(t2 * (1.f / D_) - mu * mu, 0.f) + LN_EPS_);
    #pragma unroll
    for (int i = 0; i < 4; ++i) {
        int c = threadIdx.x + i * 128;
        float y = (v[i] - mu) * inv * g[c] + bb[c];
        Y32[row * D_ + c] = y;
        Y16[row * D_ + c] = (_Float16)y;
    }
}

// ---------------- flash attention, wave-per-16-query-rows ----------------
// grid = B*HEADS*(S/64), block 128 (4 waves). K tile staged by TDM, V tile
// transposed manually (TDM has no transpose), P re-fragmented via LDS.
__global__ __launch_bounds__(128) void attn_flash(const _Float16* __restrict__ Qm,
                                                  const _Float16* __restrict__ Km,
                                                  const _Float16* __restrict__ Vm,
                                                  const int* __restrict__ masks,
                                                  _Float16* __restrict__ Oc) {
    // single block so TDM lds_addr offsets are well-defined
    __shared__ _Float16 smem[64 * 72 + 64 * 72 + 4 * 16 * 72];
    __shared__ int kmsk[64];
    _Float16 (*Kt)[72]     = (_Float16(*)[72])smem;                  // off 0
    _Float16 (*Vt)[72]     = (_Float16(*)[72])(smem + 64 * 72);      // off 9216 B
    _Float16 (*Pw)[16][72] = (_Float16(*)[16][72])(smem + 2 * 64 * 72);

    const int tid = threadIdx.x, lane = tid & 31, wave = tid >> 5;
    const int ll = lane & 15, lh = lane >> 4;
    const int wg = blockIdx.x;
    const int qb = wg & 7, head = (wg >> 3) & 7, b = wg >> 6;

    const size_t rowQ = (size_t)b * S_ + qb * 64 + wave * 16 + ll;
    const _Float16* qp = Qm + rowQ * 512 + head * DK_;
    v16h qf0 = ldfrag(qp + lh * 16);
    v16h qf1 = ldfrag(qp + 32 + lh * 16);

    int mq[8];
    #pragma unroll
    for (int r = 0; r < 8; ++r) mq[r] = masks[b * S_ + qb * 64 + wave * 16 + r + 8 * lh];

    float rmax[8], rsum[8];
    #pragma unroll
    for (int r = 0; r < 8; ++r) { rmax[r] = -3e38f; rsum[r] = 0.f; }
    v8f accO[4] = {};

    for (int kb = 0; kb < 8; ++kb) {
        const _Float16* kg = Km + ((size_t)b * S_ + kb * 64) * 512 + head * DK_;
#if HAVE_TDM
        if (wave == 0) {
            // K tile: 64 rows x 64 halves (128B row = 32 DW -> pad_interval=4)
            tdm_load_2d(kg, 0u, 64u, 64u, 512u, 4u);
        }
#else
        #pragma unroll
        for (int i = 0; i < 4; ++i) {
            int t = tid + i * 128;
            int r = t >> 3, c = (t & 7) * 8;
            *(uint4*)&Kt[r][c] = *(const uint4*)&kg[(size_t)r * 512 + c];
        }
#endif
        const _Float16* vg = Vm + ((size_t)b * S_ + kb * 64) * 512 + head * DK_;
        #pragma unroll
        for (int i = 0; i < 32; ++i) {                 // V tile, transposed into LDS
            int e = tid + i * 128;
            int d = e >> 6, kk = e & 63;
            Vt[d][kk] = vg[(size_t)kk * 512 + d];
        }
        if (tid < 64) kmsk[tid] = masks[b * S_ + kb * 64 + tid];
#if HAVE_TDM
        if (wave == 0) wait_tensorcnt0();
#else
        __builtin_prefetch(kg + 64 * 512, 0, 1);
#endif
        __syncthreads();

        float sc[4][8];
        int mk[4];
        #pragma unroll
        for (int nt = 0; nt < 4; ++nt) {               // S = Q @ K^T (scaled)
            v8f s = {};
            s = wmma_f16(qf0, ldfrag(&Kt[nt * 16 + ll][lh * 16]), s);
            s = wmma_f16(qf1, ldfrag(&Kt[nt * 16 + ll][32 + lh * 16]), s);
            mk[nt] = kmsk[nt * 16 + ll];
            #pragma unroll
            for (int r = 0; r < 8; ++r) {
                float v2 = s[r] * 0.125f;              // 1/sqrt(DK)
                sc[nt][r] = (mq[r] && mk[nt]) ? v2 : -1e8f;
            }
        }
        float al[8];
        #pragma unroll
        for (int r = 0; r < 8; ++r) {                  // online softmax row stats
            float m = fmaxf(fmaxf(sc[0][r], sc[1][r]), fmaxf(sc[2][r], sc[3][r]));
            #pragma unroll
            for (int o = 1; o < 16; o <<= 1) m = fmaxf(m, __shfl_xor(m, o, 16));
            float nm = fmaxf(rmax[r], m);
            al[r] = __expf(rmax[r] - nm);
            rmax[r] = nm;
            float ps = 0.f;
            #pragma unroll
            for (int nt = 0; nt < 4; ++nt) {
                float p = (mq[r] && mk[nt]) ? __expf(sc[nt][r] - nm) : 0.f;
                sc[nt][r] = p;
                ps += p;
            }
            #pragma unroll
            for (int o = 1; o < 16; o <<= 1) ps += __shfl_xor(ps, o, 16);
            rsum[r] = rsum[r] * al[r] + ps;
            #pragma unroll
            for (int dt = 0; dt < 4; ++dt) accO[dt][r] *= al[r];
        }
        #pragma unroll
        for (int nt = 0; nt < 4; ++nt)                 // P -> LDS, re-fragment
            #pragma unroll
            for (int r = 0; r < 8; ++r)
                Pw[wave][r + 8 * lh][nt * 16 + ll] = (_Float16)sc[nt][r];
        __syncthreads();
        #pragma unroll
        for (int kc = 0; kc < 2; ++kc) {               // O += P @ V
            v16h pf = ldfrag(&Pw[wave][ll][kc * 32 + lh * 16]);
            #pragma unroll
            for (int dt = 0; dt < 4; ++dt) {
                v16h vf = ldfrag(&Vt[dt * 16 + ll][kc * 32 + lh * 16]);
                accO[dt] = wmma_f16(pf, vf, accO[dt]);
            }
        }
        __syncthreads();
    }
    #pragma unroll
    for (int dt = 0; dt < 4; ++dt)
        #pragma unroll
        for (int r = 0; r < 8; ++r) {
            float o = accO[dt][r] / fmaxf(rsum[r], 1e-20f);
            size_t row = (size_t)b * S_ + qb * 64 + wave * 16 + r + 8 * lh;
            Oc[row * 512 + head * DK_ + dt * 16 + ll] = (_Float16)o;
        }
}

// ---------------- residual + rowmask + LN2, in-place on d_out ----------------
__global__ __launch_bounds__(128) void final_ln2(float* __restrict__ Y,
                                                 const float* __restrict__ O,
                                                 const int* __restrict__ masks,
                                                 const float* __restrict__ g,
                                                 const float* __restrict__ bb) {
    __shared__ float red[4];
    size_t row = blockIdx.x;
    int bidx = (int)(row / S_), s = (int)(row % S_);
    float rm = masks[bidx * S_ + s] ? 1.f : 0.f;
    float v[4], s1 = 0.f, s2 = 0.f;
    #pragma unroll
    for (int i = 0; i < 4; ++i) {
        int c = threadIdx.x + i * 128;
        v[i] = Y[row * D_ + c] + O[row * D_ + c] * rm;
        s1 += v[i]; s2 += v[i] * v[i];
    }
    float t1 = blk_sum128(s1, red);
    float t2 = blk_sum128(s2, red);
    float mu  = t1 * (1.f / D_);
    float inv = rsqrtf(fmaxf(t2 * (1.f / D_) - mu * mu, 0.f) + LN_EPS_);
    #pragma unroll
    for (int i = 0; i < 4; ++i) {
        int c = threadIdx.x + i * 128;
        Y[row * D_ + c] = (v[i] - mu) * inv * g[c] + bb[c];
    }
}

// ---------------- host orchestration ----------------
extern "C" void kernel_launch(void* const* d_in, const int* in_sizes, int n_in,
                              void* d_out, int out_size, void* d_ws, size_t ws_size,
                              hipStream_t stream) {
    (void)in_sizes; (void)n_in; (void)out_size; (void)ws_size;
    const float* splits = (const float*)d_in[0];
    const float* w_ih_f = (const float*)d_in[1];
    const float* w_hh_f = (const float*)d_in[2];
    const float* b_ih_f = (const float*)d_in[3];
    const float* b_hh_f = (const float*)d_in[4];
    const float* w_ih_b = (const float*)d_in[5];
    const float* w_hh_b = (const float*)d_in[6];
    const float* b_ih_b = (const float*)d_in[7];
    const float* b_hh_b = (const float*)d_in[8];
    const float* ln1_g  = (const float*)d_in[9];
    const float* ln1_b  = (const float*)d_in[10];
    const float* wq = (const float*)d_in[11];
    const float* bq = (const float*)d_in[12];
    const float* wk = (const float*)d_in[13];
    const float* bk = (const float*)d_in[14];
    const float* wv = (const float*)d_in[15];
    const float* bv = (const float*)d_in[16];
    const float* wo = (const float*)d_in[17];
    const float* bo = (const float*)d_in[18];
    const float* ln2_g = (const float*)d_in[19];
    const float* ln2_b = (const float*)d_in[20];
    const int* lengths = (const int*)d_in[21];
    const int* masks   = (const int*)d_in[22];

    size_t off = 0;
    auto alloc = [&](size_t bytes) {
        void* p = (char*)d_ws + off;
        off += (bytes + 255) & ~(size_t)255;
        return p;
    };
    _Float16* wihf16 = (_Float16*)alloc((size_t)G3_ * D_ * 2);
    _Float16* wihb16 = (_Float16*)alloc((size_t)G3_ * D_ * 2);
    _Float16* whhf16 = (_Float16*)alloc((size_t)G3_ * H2_ * 2);
    _Float16* whhb16 = (_Float16*)alloc((size_t)G3_ * H2_ * 2);
    _Float16* wq16   = (_Float16*)alloc((size_t)D_ * D_ * 2);
    _Float16* wk16   = (_Float16*)alloc((size_t)D_ * D_ * 2);
    _Float16* wv16   = (_Float16*)alloc((size_t)D_ * D_ * 2);
    _Float16* wo16   = (_Float16*)alloc((size_t)D_ * D_ * 2);
    _Float16* x16    = (_Float16*)alloc((size_t)BS_ * D_ * 2);   // reused: attn concat
    _Float16* xgf16  = (_Float16*)alloc((size_t)BS_ * G3_ * 2);  // reused: Q/K/V (with xgb)
    _Float16* xgb16  = (_Float16*)alloc((size_t)BS_ * G3_ * 2);
    float*    gruout = (float*)alloc((size_t)BS_ * D_ * 4);      // reused: O-proj f32
    _Float16* ln116  = (_Float16*)alloc((size_t)BS_ * D_ * 2);

    _Float16* q16 = xgf16;                        // 3 * BS*512 halves == 2 * BS*768 halves
    _Float16* k16 = q16 + (size_t)BS_ * D_;
    _Float16* v16 = k16 + (size_t)BS_ * D_;
    _Float16* concat16 = x16;
    float* o32 = gruout;
    float* out = (float*)d_out;                   // [0,BS*D): outputs ; tail: hidden

    // 1) cast everything that feeds WMMA to f16
    cast16<<<512, 256, 0, stream>>>(splits, x16, BS_ * D_);
    cast16<<<128, 256, 0, stream>>>(w_ih_f, wihf16, G3_ * D_);
    cast16<<<128, 256, 0, stream>>>(w_ih_b, wihb16, G3_ * D_);
    cast16<<<128, 256, 0, stream>>>(w_hh_f, whhf16, G3_ * H2_);
    cast16<<<128, 256, 0, stream>>>(w_hh_b, whhb16, G3_ * H2_);
    cast16<<<128, 256, 0, stream>>>(wq, wq16, D_ * D_);
    cast16<<<128, 256, 0, stream>>>(wk, wk16, D_ * D_);
    cast16<<<128, 256, 0, stream>>>(wv, wv16, D_ * D_);
    cast16<<<128, 256, 0, stream>>>(wo, wo16, D_ * D_);

    // 2) input-gate projections (both directions): [16384,768] = x @ w_ih^T + b_ih
    gemm128x64<true><<<dim3(BS_ / 128, G3_ / 64), 256, 0, stream>>>(x16, wihf16, b_ih_f, xgf16, D_, G3_);
    gemm128x64<true><<<dim3(BS_ / 128, G3_ / 64), 256, 0, stream>>>(x16, wihb16, b_ih_b, xgb16, D_, G3_);

    // 3) sequential bidirectional GRU (persistent, WMMA recurrence)
    gru_scan<<<4, 512, 0, stream>>>(xgf16, xgb16, whhf16, whhb16, b_hh_f, b_hh_b,
                                    masks, lengths, gruout, out);

    // 4) LN1 -> d_out (f32 scratch) + f16 copy for GEMMs
    ln_write<<<BS_, 128, 0, stream>>>(gruout, ln1_g, ln1_b, out, ln116);

    // 5) Q/K/V projections
    gemm128x64<true><<<dim3(BS_ / 128, D_ / 64), 256, 0, stream>>>(ln116, wq16, bq, q16, D_, D_);
    gemm128x64<true><<<dim3(BS_ / 128, D_ / 64), 256, 0, stream>>>(ln116, wk16, bk, k16, D_, D_);
    gemm128x64<true><<<dim3(BS_ / 128, D_ / 64), 256, 0, stream>>>(ln116, wv16, bv, v16, D_, D_);

    // 6) masked flash attention
    attn_flash<<<B_ * HEADS_ * (S_ / 64), 128, 0, stream>>>(q16, k16, v16, masks, concat16);

    // 7) output projection (f32 out)
    gemm128x64<false><<<dim3(BS_ / 128, D_ / 64), 256, 0, stream>>>(concat16, wo16, bo, o32, D_, D_);

    // 8) residual (row-masked) + LN2, in place on d_out
    final_ln2<<<BS_, 128, 0, stream>>>(out, o32, masks, ln2_g, ln2_b);
}
